// MultiHeadAttention_9277129359942
// MI455X (gfx1250) — matrix operand order repaired and checked by reading, not compiled, not run on hardware
//
#include <hip/hip_runtime.h>
#include <hip/hip_bf16.h>
#include <stdint.h>

// ---------------------------------------------------------------------------
// MHA forward for MI455X (gfx1250), wave32.
// bf16 WMMA + CDNA5 async global->LDS staging (double-buffered, 64-wide
// chunks) + VALU permlane16 softmax reductions (co-execute with WMMA).
// ---------------------------------------------------------------------------

typedef __bf16 bf16;
typedef __attribute__((ext_vector_type(16))) __bf16 v16bf;
typedef __attribute__((ext_vector_type(8)))  __bf16 v8bf;
typedef __attribute__((ext_vector_type(8)))  float   v8f;

#define D_MODEL 1024
#define NHEAD   16
#define HD      64
#define LOG2E   1.44269504088896340736f

__device__ __forceinline__ unsigned short f2bf_u(float f) {
  union { float f; uint32_t u; } x; x.f = f;
  uint32_t r = x.u + 0x7FFFu + ((x.u >> 16) & 1u);   // round-nearest-even
  return (unsigned short)(r >> 16);
}
__device__ __forceinline__ bf16 f2bf(float f) {
  unsigned short h = f2bf_u(f);
  bf16 b; __builtin_memcpy(&b, &h, 2); return b;
}

union FragA { v16bf v; v8bf h[2]; };

// ---- CDNA5 async global->LDS copy (ASYNCcnt path) -------------------------
__device__ __forceinline__ void async_b128(void* ldsPtr, const void* gBase,
                                           uint32_t voffBytes) {
  uint32_t loff = (uint32_t)(uintptr_t)ldsPtr;   // LDS address taken from low 32 bits
  asm volatile("global_load_async_to_lds_b128 %0, %1, %2"
               :: "v"(loff), "v"(voffBytes), "s"(gBase) : "memory");
}
__device__ __forceinline__ void wait_async0() {
  asm volatile("s_wait_asynccnt 0" ::: "memory");
}

// ---- VALU xor-shuffle within 16-lane rows via v_permlane16_b32 ------------
// nibble j of {sel_hi,sel_lo} = j ^ mask  (per 16-lane row; wave32 = 2 rows)
template<uint32_t SEL_LO, uint32_t SEL_HI>
__device__ __forceinline__ float permlane16f(float v) {
  int x = __builtin_bit_cast(int, v);
  int r = __builtin_amdgcn_permlane16(x, x, (int)SEL_LO, (int)SEL_HI, false, false);
  return __builtin_bit_cast(float, r);
}
__device__ __forceinline__ float rowmax16(float t) {
  t = fmaxf(t, permlane16f<0x67452301u, 0xEFCDAB89u>(t));  // xor 1
  t = fmaxf(t, permlane16f<0x54761032u, 0xDCFE98BAu>(t));  // xor 2
  t = fmaxf(t, permlane16f<0x32107654u, 0xBA98FEDCu>(t));  // xor 4
  t = fmaxf(t, permlane16f<0xFEDCBA98u, 0x76543210u>(t));  // xor 8
  return t;
}
__device__ __forceinline__ float rowsum16(float t) {
  t += permlane16f<0x67452301u, 0xEFCDAB89u>(t);
  t += permlane16f<0x54761032u, 0xDCFE98BAu>(t);
  t += permlane16f<0x32107654u, 0xBA98FEDCu>(t);
  t += permlane16f<0xFEDCBA98u, 0x76543210u>(t);
  return t;
}

// ---------------------------------------------------------------------------
// Stage 0: f32 -> bf16 conversion (grid-stride)
// ---------------------------------------------------------------------------
__global__ void cvt_f32_bf16(const float* __restrict__ in, bf16* __restrict__ out, int n) {
  int i = blockIdx.x * blockDim.x + threadIdx.x;
  int stride = gridDim.x * blockDim.x;
  for (; i < n; i += stride) out[i] = f2bf(in[i]);
}

// ---------------------------------------------------------------------------
// Stage 1/3: C[r,c] = sum_k A[r,k] * W[c,k] + bias[c]   (torch Linear)
// Block = 8 waves sharing one 64-column W strip; 64-wide K chunks (64x64 bf16
// = 8KB) async-DMA'd into double-buffered LDS. 8 WMMAs per iteration/wave.
// MODE 0: out bf16 [B,H,T,64]; MODE 1: out bf16 [B,H,64,S]; MODE 2: f32 [M,N].
// ---------------------------------------------------------------------------
template<int MODE>
__global__ __launch_bounds__(256, 2) void gemm_wmma(
    const bf16* __restrict__ A, const bf16* __restrict__ W,
    const float* __restrict__ bias, void* __restrict__ out,
    int M, int K, int N, int Tlen)
{
  __shared__ __align__(64) bf16 ldsW[2][64 * 64];   // 2 x 8KB double buffer

  const int tid    = threadIdx.x;
  const int lane   = tid & 31;
  const int wave   = tid >> 5;
  const int numCt  = N / 64;
  const int ct     = blockIdx.x % numCt;            // block-uniform W strip
  const int rt     = (blockIdx.x / numCt) * 8 + wave;

  const int half16 = lane >> 4;
  const int l16    = lane & 15;
  const int koff   = half16 * 16;                   // contiguous k split (matches B)
  const int arow   = rt * 16 + l16;

  // staging role: W row (= output col) 0..63, 32B sub-chunk 0..3
  const int wcol = tid >> 2;
  const int wc16 = (tid & 3) * 16;                  // element offset in K-chunk

  v8f acc[4];
  #pragma unroll
  for (int s = 0; s < 4; ++s)
    #pragma unroll
    for (int i = 0; i < 8; ++i) acc[s][i] = 0.f;

  // prologue: fill buffer 0 with W[:, 0..63]
  async_b128(&ldsW[0][wcol * 64 + wc16],     W, (uint32_t)(((ct*64+wcol)*K + wc16) * 2));
  async_b128(&ldsW[0][wcol * 64 + wc16 + 8], W, (uint32_t)(((ct*64+wcol)*K + wc16 + 8) * 2));
  wait_async0();
  __syncthreads();

  const bf16* Arow = A + (size_t)arow * K;
  for (int kk = 0; kk < K; kk += 64) {
    const int cur = (kk >> 6) & 1;
    if (kk + 64 < K) {   // DMA next W chunk while computing on current
      const uint32_t so = (uint32_t)(((ct*64+wcol)*K + kk + 64 + wc16) * 2);
      async_b128(&ldsW[cur ^ 1][wcol * 64 + wc16],     W, so);
      async_b128(&ldsW[cur ^ 1][wcol * 64 + wc16 + 8], W, so + 16);
    }

    FragA af[2];
    #pragma unroll
    for (int kp = 0; kp < 2; ++kp) {
      af[kp].h[0] = *(const v8bf*)(Arow + kk + kp * 32 + koff);
      af[kp].h[1] = *(const v8bf*)(Arow + kk + kp * 32 + 8 + koff);
    }
    #pragma unroll
    for (int s = 0; s < 4; ++s) {
      #pragma unroll
      for (int kp = 0; kp < 2; ++kp) {
        const v16bf bfrag =
            *(const v16bf*)&ldsW[cur][(s * 16 + l16) * 64 + kp * 32 + half16 * 16];
        acc[s] = __builtin_amdgcn_wmma_f32_16x16x32_bf16(
            false, af[kp].v, false, bfrag, (short)0, acc[s], false, false);
      }
    }
    wait_async0();
    __syncthreads();
  }

  // Epilogue: C layout -> VGPR i holds (row i + 8*half16, col l16)
  #pragma unroll
  for (int s = 0; s < 4; ++s) {
    const int c  = ct * 64 + s * 16 + l16;
    const float bv = bias[c];
    #pragma unroll
    for (int i = 0; i < 8; ++i) {
      const int r = rt * 16 + i + half16 * 8;
      const float val = acc[s][i] + bv;
      if (MODE == 2) {
        ((float*)out)[(size_t)r * N + c] = val;
      } else {
        const int b = r / Tlen, t = r % Tlen;
        const int h = c >> 6,  d = c & 63;
        bf16* o = (bf16*)out;
        if (MODE == 0)
          o[((size_t)(b * NHEAD + h) * Tlen + t) * HD + d] = f2bf(val);
        else
          o[((size_t)(b * NHEAD + h) * HD + d) * Tlen + t] = f2bf(val);
      }
    }
  }
}

// ---------------------------------------------------------------------------
// Stage 2: flash attention. Block = 8 waves sharing one (b,h); each wave owns
// 16 query rows. 64-wide S chunks: K tile (64x64) and V^T tile (64x64)
// async-DMA'd once per block into double-buffered LDS. Per iteration/wave:
// 8 score WMMAs, one permlane max-reduction, per-lane partial row-sums
// (reduced once at the end), P via per-wave LDS, 8 context WMMAs.
// ---------------------------------------------------------------------------
__global__ __launch_bounds__(256, 2) void attn_wmma(
    const bf16* __restrict__ Qh, const bf16* __restrict__ Kh,
    const bf16* __restrict__ VhT, bf16* __restrict__ Ctx,
    int Tn, int Sn)
{
  __shared__ __align__(64) bf16 ldsK[2][64 * 64];   // 2 x 8KB
  __shared__ __align__(64) bf16 ldsV[2][64 * 64];   // 2 x 8KB
  __shared__ __align__(64) bf16 ldsP[8][16 * 64];   // 2KB per wave

  const int tid    = threadIdx.x;
  const int lane   = tid & 31;
  const int wave   = tid >> 5;
  const int bh     = blockIdx.x >> 4;               // 16 blocks per (b,h)
  const int tt     = (blockIdx.x & 15) * 8 + wave;  // wave's 16-row tile
  const int b      = bh / NHEAD, h = bh % NHEAD;
  const int half16 = lane >> 4, l16 = lane & 15;

  const bf16* Qp = Qh  + (size_t)bh * Tn * HD;
  const bf16* Kp = Kh  + (size_t)bh * Sn * HD;
  const bf16* Vp = VhT + (size_t)bh * HD * Sn;
  bf16* ldsp = &ldsP[wave][0];

  // staging roles: K tile is 8KB contiguous; V tile is 64 rows x 128B
  const int ke16 = tid * 16;                        // element offset in K tile
  const int vrow = tid >> 2, vc16 = (tid & 3) * 16;

  // Q fragments (hd=64 -> two K=32 steps), loaded once.
  FragA qa[2];
  {
    const bf16* qrow = Qp + (size_t)(tt * 16 + l16) * HD;
    const int qoff = half16 * 16;
    #pragma unroll
    for (int kp = 0; kp < 2; ++kp) {
      qa[kp].h[0] = *(const v8bf*)(qrow + kp * 32 + qoff);
      qa[kp].h[1] = *(const v8bf*)(qrow + kp * 32 + 8 + qoff);
    }
  }

  float m[8], l[8];   // l = per-lane partial row-sum (reduced at the end)
  #pragma unroll
  for (int i = 0; i < 8; ++i) { m[i] = -1e30f; l[i] = 0.f; }
  v8f accO[4];
  #pragma unroll
  for (int d = 0; d < 4; ++d)
    #pragma unroll
    for (int i = 0; i < 8; ++i) accO[d][i] = 0.f;

  const float scale = 0.125f;  // 1/sqrt(64)

  // prologue: fill buffer 0 with s0 = 0
  async_b128(&ldsK[0][ke16],     Kp, (uint32_t)(ke16 * 2));
  async_b128(&ldsK[0][ke16 + 8], Kp, (uint32_t)(ke16 * 2 + 16));
  async_b128(&ldsV[0][vrow * 64 + vc16],     Vp, (uint32_t)((vrow * (size_t)Sn + vc16) * 2));
  async_b128(&ldsV[0][vrow * 64 + vc16 + 8], Vp, (uint32_t)((vrow * (size_t)Sn + vc16 + 8) * 2));
  wait_async0();
  __syncthreads();

  for (int s0 = 0; s0 < Sn; s0 += 64) {
    const int cur = (s0 >> 6) & 1;
    if (s0 + 64 < Sn) {   // DMA next K/V chunk while computing on current
      const uint32_t ko = (uint32_t)(((s0 + 64) * HD + ke16) * 2);
      async_b128(&ldsK[cur ^ 1][ke16],     Kp, ko);
      async_b128(&ldsK[cur ^ 1][ke16 + 8], Kp, ko + 16);
      const uint32_t vo = (uint32_t)((vrow * (size_t)Sn + s0 + 64 + vc16) * 2);
      async_b128(&ldsV[cur ^ 1][vrow * 64 + vc16],     Vp, vo);
      async_b128(&ldsV[cur ^ 1][vrow * 64 + vc16 + 8], Vp, vo + 16);
    }

    // ---- scores: four 16x16 tiles over s0..s0+63 ----
    v8f sacc[4];
    #pragma unroll
    for (int sh = 0; sh < 4; ++sh)
      #pragma unroll
      for (int i = 0; i < 8; ++i) sacc[sh][i] = 0.f;
    #pragma unroll
    for (int sh = 0; sh < 4; ++sh) {
      #pragma unroll
      for (int kp = 0; kp < 2; ++kp) {
        const v16bf bfrag = *(const v16bf*)
            &ldsK[cur][(sh * 16 + l16) * 64 + kp * 32 + half16 * 16];
        sacc[sh] = __builtin_amdgcn_wmma_f32_16x16x32_bf16(
            false, qa[kp].v, false, bfrag, (short)0, sacc[sh], false, false);
      }
    }

    // ---- online softmax (VALU permlane reductions, co-execute w/ WMMA) ----
    float corr[8];
    #pragma unroll
    for (int i = 0; i < 8; ++i) {
      float t = fmaxf(fmaxf(sacc[0][i], sacc[1][i]),
                      fmaxf(sacc[2][i], sacc[3][i])) * scale;
      t = rowmax16(t);
      const float mn = fmaxf(m[i], t);
      corr[i] = exp2f((m[i] - mn) * LOG2E);
      float ps = 0.f;
      #pragma unroll
      for (int sh = 0; sh < 4; ++sh) {
        const float p = exp2f((sacc[sh][i] * scale - mn) * LOG2E);
        sacc[sh][i] = p;
        ps += p;
      }
      l[i] = l[i] * corr[i] + ps;   // per-lane partial; reduced in epilogue
      m[i] = mn;
    }
    #pragma unroll
    for (int d = 0; d < 4; ++d)
      #pragma unroll
      for (int i = 0; i < 8; ++i) accO[d][i] *= corr[i];

    // ---- P: C-layout -> row-major per-wave LDS -> two A-fragments ----
    #pragma unroll
    for (int i = 0; i < 8; ++i) {
      const int row = i + half16 * 8;
      #pragma unroll
      for (int sh = 0; sh < 4; ++sh)
        ldsp[row * 64 + sh * 16 + l16] = f2bf(sacc[sh][i]);
    }
    FragA pa[2];   // same-wave DS ops are in-order
    #pragma unroll
    for (int kp = 0; kp < 2; ++kp) {
      pa[kp].h[0] = *(const v8bf*)(ldsp + l16 * 64 + kp * 32 + half16 * 16);
      pa[kp].h[1] = *(const v8bf*)(ldsp + l16 * 64 + kp * 32 + 8 + half16 * 16);
    }

    // ---- ctx += P V (B column d = V^T row d, from LDS) ----
    #pragma unroll
    for (int kp = 0; kp < 2; ++kp) {
      #pragma unroll
      for (int d = 0; d < 4; ++d) {
        const v16bf bfrag = *(const v16bf*)
            &ldsV[cur][(d * 16 + l16) * 64 + kp * 32 + half16 * 16];
        accO[d] = __builtin_amdgcn_wmma_f32_16x16x32_bf16(
            false, pa[kp].v, false, bfrag, (short)0, accO[d], false, false);
      }
    }

    wait_async0();
    __syncthreads();
  }

  // ---- final row-sum reduction, normalize, emit ctx [B*T, D] bf16 ----
  float inv[8];
  #pragma unroll
  for (int i = 0; i < 8; ++i) inv[i] = 1.f / rowsum16(l[i]);
  #pragma unroll
  for (int d = 0; d < 4; ++d) {
    const int c = h * HD + d * 16 + l16;
    #pragma unroll
    for (int i = 0; i < 8; ++i) {
      const int r = b * Tn + tt * 16 + i + half16 * 8;
      Ctx[(size_t)r * D_MODEL + c] = f2bf(accO[d][i] * inv[i]);
    }
  }
}

// ---------------------------------------------------------------------------
// Host-side orchestration
// ---------------------------------------------------------------------------
extern "C" void kernel_launch(void* const* d_in, const int* in_sizes, int n_in,
                              void* d_out, int out_size, void* d_ws, size_t ws_size,
                              hipStream_t stream)
{
  const float* q_in = (const float*)d_in[0];
  const float* k_in = (const float*)d_in[1];
  const float* v_in = (const float*)d_in[2];
  const float* Wq   = (const float*)d_in[3];
  const float* bq   = (const float*)d_in[4];
  const float* Wk   = (const float*)d_in[5];
  const float* bk   = (const float*)d_in[6];
  const float* Wv   = (const float*)d_in[7];
  const float* bv   = (const float*)d_in[8];
  const float* Wo   = (const float*)d_in[9];
  const float* bo   = (const float*)d_in[10];

  const int B = 2, T = 2048, S = 2048;
  const int M = B * T;            // 4096
  const int K = D_MODEL, N = D_MODEL;

  char* ws = (char*)d_ws;
  size_t off = 0;
  auto alloc = [&](size_t bytes) -> void* {
    void* p = ws + off;
    off = (off + bytes + 255) & ~(size_t)255;
    return p;
  };
  bf16* xq  = (bf16*)alloc((size_t)M * K * 2);
  bf16* xk  = (bf16*)alloc((size_t)M * K * 2);
  bf16* xv  = (bf16*)alloc((size_t)M * K * 2);
  bf16* wqb = (bf16*)alloc((size_t)N * K * 2);
  bf16* wkb = (bf16*)alloc((size_t)N * K * 2);
  bf16* wvb = (bf16*)alloc((size_t)N * K * 2);
  bf16* wob = (bf16*)alloc((size_t)N * K * 2);
  bf16* qh  = (bf16*)alloc((size_t)M * K * 2);   // [B,H,T,64]
  bf16* kh  = (bf16*)alloc((size_t)M * K * 2);   // [B,H,S,64]
  bf16* vt  = (bf16*)alloc((size_t)M * K * 2);   // [B,H,64,S]
  bf16* ctx = (bf16*)alloc((size_t)M * K * 2);   // [B*T, D]

  // Stage 0: conversions
  cvt_f32_bf16<<<512, 256, 0, stream>>>(q_in, xq, M * K);
  cvt_f32_bf16<<<512, 256, 0, stream>>>(k_in, xk, M * K);
  cvt_f32_bf16<<<512, 256, 0, stream>>>(v_in, xv, M * K);
  cvt_f32_bf16<<<512, 256, 0, stream>>>(Wq, wqb, N * K);
  cvt_f32_bf16<<<512, 256, 0, stream>>>(Wk, wkb, N * K);
  cvt_f32_bf16<<<512, 256, 0, stream>>>(Wv, wvb, N * K);
  cvt_f32_bf16<<<512, 256, 0, stream>>>(Wo, wob, N * K);

  // Stage 1: QKV projections (block = 8 waves sharing a 64-col W strip)
  const int gemmBlocks = (M / 16 / 8) * (N / 64);   // 512
  gemm_wmma<0><<<gemmBlocks, 256, 0, stream>>>(xq, wqb, bq, qh, M, K, N, T);
  gemm_wmma<0><<<gemmBlocks, 256, 0, stream>>>(xk, wkb, bk, kh, M, K, N, S);
  gemm_wmma<1><<<gemmBlocks, 256, 0, stream>>>(xv, wvb, bv, vt, M, K, N, S);

  // Stage 2: flash attention (block = 8 waves sharing one (b,h))
  const int attnBlocks = (B * NHEAD) * (T / 16) / 8;  // 512
  attn_wmma<<<attnBlocks, 256, 0, stream>>>(qh, kh, vt, ctx, T, S);

  // Stage 3: output projection to f32 d_out
  gemm_wmma<2><<<gemmBlocks, 256, 0, stream>>>(ctx, wob, bo, d_out, M, K, N, T);
}
